// SpikingNeuronLayer_4681514352744
// MI455X (gfx1250) — matrix operand order repaired and checked by reading, not compiled, not run on hardware
//
#include <hip/hip_runtime.h>

typedef __attribute__((ext_vector_type(16))) __bf16 v16bf;
typedef __attribute__((ext_vector_type(8)))  float  v8f;

#define B_    32
#define T_    1024
#define IN_   512
#define HID_  512
#define HT_   64           // h columns per block
#define CH_   32           // timesteps per chunk (2 WMMA M-tiles)
#define NCH_  (T_ / CH_)
#define WS_   (IN_ + 8)    // padded LDS row stride for W tiles (u16 elems)
#define XS_   (IN_ + 8)    // padded LDS row stride for x tiles (u16 elems)
#define ZS_   (HT_ + 4)    // padded LDS row stride for Z tile (floats)
#define ALPHA_ 0.951229424500714f   // exp(-1/20)

__device__ __forceinline__ unsigned short f2bf(float f) {
  unsigned u = __float_as_uint(f);
  u += 0x7FFFu + ((u >> 16) & 1u);           // round-to-nearest-even
  return (unsigned short)(u >> 16);
}
__device__ __forceinline__ float bf2f(unsigned short h) {
  return __uint_as_float(((unsigned)h) << 16);
}

// Split fp32 W into bf16 hi/lo pair (bf16x3 trick gives ~fp32 GEMM accuracy).
__global__ void snn_wsplit_kernel(const float* __restrict__ W,
                                  unsigned short* __restrict__ whi,
                                  unsigned short* __restrict__ wlo, int n) {
  int i = blockIdx.x * blockDim.x + threadIdx.x;
  if (i < n) {
    float w = W[i];
    unsigned short h = f2bf(w);
    whi[i] = h;
    wlo[i] = f2bf(w - bf2f(h));
  }
}

union FragAB { v16bf v; uint4 q[2]; };

__global__ void __launch_bounds__(256)
snn_fused_kernel(const float* __restrict__ x,
                 const unsigned short* __restrict__ whi_g,
                 const unsigned short* __restrict__ wlo_g,
                 const float* __restrict__ bias_g,
                 const float* __restrict__ thr_g,
                 float* __restrict__ out) {
  extern __shared__ unsigned short smem[];
  unsigned short* whi_s = smem;                    // HT_*WS_
  unsigned short* wlo_s = whi_s + HT_ * WS_;       // HT_*WS_
  unsigned short* xhi_s = wlo_s + HT_ * WS_;       // CH_*XS_
  unsigned short* xlo_s = xhi_s + CH_ * XS_;       // CH_*XS_
  float*          z_s   = (float*)(xlo_s + CH_ * XS_);  // CH_*ZS_

  const int tid  = threadIdx.x;
  const int lane = tid & 31;
  const int wv   = tid >> 5;            // wave 0..7
  const int mt   = wv >> 2;             // M-tile (timestep group 0/1)
  const int nt   = wv & 3;              // N-tile (h subtile 0..3)
  const int h0   = blockIdx.x * HT_;    // h-tile base
  const int b    = blockIdx.y;          // batch

  // ---- stage W hi/lo tile into LDS (resident for all 1024 timesteps) ----
  {
    const uint4* whi4 = (const uint4*)whi_g;
    const uint4* wlo4 = (const uint4*)wlo_g;
    for (int i = tid; i < HT_ * (IN_ / 8); i += 256) {
      int row = i >> 6;                 // IN_/8 == 64 uint4 per row
      int col = i & 63;
      *(uint4*)(whi_s + row * WS_ + col * 8) = whi4[(h0 + row) * (IN_ / 8) + col];
      *(uint4*)(wlo_s + row * WS_ + col * 8) = wlo4[(h0 + row) * (IN_ / 8) + col];
    }
  }

  // persistent scan state: one h column per thread (tid < HT_)
  float mem = 0.f, bia = 0.f, thr = 0.f;
  if (tid < HT_) {
    bia = bias_g[h0 + tid];
    thr = thr_g[h0 + tid];
  }

  const float4* xg4 = (const float4*)(x + ((size_t)b * T_) * IN_);
  // WMMA fragment geometry (ISA §7.12.2, 16-bit A 16x32 / B 32x16 / f32 C 16x16)
  const int abase = (lane < 16) ? 0 : 8;    // A: K sub-offset per half-wave
  const int bbase = (lane < 16) ? 0 : 8;    // B: K sub-offset per half-wave
  const int trow  = (mt << 4) + (lane & 15);// A row (timestep within chunk)
  const int hrow  = (nt << 4) + (lane & 15);// B column (h within tile)
  const int mbase = (mt << 4) + ((lane < 16) ? 0 : 8);  // C/D: Z row = mbase + r

  for (int chunk = 0; chunk < NCH_; ++chunk) {
    const int t0 = chunk * CH_;

    // ---- stage x[b, t0..t0+31, :] into LDS, split bf16 hi/lo ----
    for (int i = tid; i < CH_ * IN_ / 4; i += 256) {
      int t = i >> 7;
      int k = (i & 127) << 2;
      float4 v = xg4[(size_t)chunk * (CH_ * IN_ / 4) + i];
      unsigned short* ph = xhi_s + t * XS_ + k;
      unsigned short* pl = xlo_s + t * XS_ + k;
      unsigned short a0 = f2bf(v.x), a1 = f2bf(v.y), a2 = f2bf(v.z), a3 = f2bf(v.w);
      ph[0] = a0; ph[1] = a1; ph[2] = a2; ph[3] = a3;
      pl[0] = f2bf(v.x - bf2f(a0)); pl[1] = f2bf(v.y - bf2f(a1));
      pl[2] = f2bf(v.z - bf2f(a2)); pl[3] = f2bf(v.w - bf2f(a3));
    }
    // warm next 64KB x chunk (global_prefetch_b8), 2 x 128B per thread
    if (chunk + 1 < NCH_) {
      const char* nx = (const char*)(xg4 + (size_t)(chunk + 1) * (CH_ * IN_ / 4));
      __builtin_prefetch(nx + tid * 256, 0, 0);
      __builtin_prefetch(nx + tid * 256 + 128, 0, 0);
    }
    __syncthreads();

    // ---- GEMM: Z[32 x 64] = Xchunk @ Wtile^T via bf16x3 WMMA ----
    v8f acc = {};
    const unsigned short* xh_row = xhi_s + trow * XS_;
    const unsigned short* xl_row = xlo_s + trow * XS_;
    const unsigned short* wh_row = whi_s + hrow * WS_;
    const unsigned short* wl_row = wlo_s + hrow * WS_;
#pragma unroll 4
    for (int k0 = 0; k0 < IN_; k0 += 32) {
      FragAB ah, al, bh, bl;
      ah.q[0] = *(const uint4*)(xh_row + k0 + abase);
      ah.q[1] = *(const uint4*)(xh_row + k0 + abase + 16);
      al.q[0] = *(const uint4*)(xl_row + k0 + abase);
      al.q[1] = *(const uint4*)(xl_row + k0 + abase + 16);
      bh.q[0] = *(const uint4*)(wh_row + k0 + bbase);
      bh.q[1] = *(const uint4*)(wh_row + k0 + bbase + 16);
      bl.q[0] = *(const uint4*)(wl_row + k0 + bbase);
      bl.q[1] = *(const uint4*)(wl_row + k0 + bbase + 16);
      acc = __builtin_amdgcn_wmma_f32_16x16x32_bf16(false, ah.v, false, bh.v,
                                                    (short)0, acc, false, false);
      acc = __builtin_amdgcn_wmma_f32_16x16x32_bf16(false, ah.v, false, bl.v,
                                                    (short)0, acc, false, false);
      acc = __builtin_amdgcn_wmma_f32_16x16x32_bf16(false, al.v, false, bh.v,
                                                    (short)0, acc, false, false);
    }
#pragma unroll
    for (int r = 0; r < 8; ++r)
      z_s[(mbase + r) * ZS_ + hrow] = acc[r];
    __syncthreads();

    // ---- sequential LIF scan over the 32 timesteps of this chunk ----
    if (tid < HT_) {
      size_t obase = ((size_t)b * T_ + t0) * HID_ + h0 + tid;
#pragma unroll
      for (int t = 0; t < CH_; ++t) {
        float z = z_s[t * ZS_ + tid] + bia;
        mem = ALPHA_ * mem + z;
        float spike = (mem >= thr) ? 1.f : 0.f;
        out[obase + (size_t)t * HID_] = spike;
        mem = mem * (1.f - spike);
      }
    }
  }

  // mem_final, concatenated after spikes in d_out
  if (tid < HT_)
    out[(size_t)B_ * T_ * HID_ + (size_t)b * HID_ + h0 + tid] = mem;
}

extern "C" void kernel_launch(void* const* d_in, const int* in_sizes, int n_in,
                              void* d_out, int out_size, void* d_ws, size_t ws_size,
                              hipStream_t stream) {
  (void)in_sizes; (void)n_in; (void)out_size; (void)ws_size;
  const float* x    = (const float*)d_in[0];
  const float* W    = (const float*)d_in[1];
  const float* bias = (const float*)d_in[2];
  const float* thr  = (const float*)d_in[3];
  float* out = (float*)d_out;

  unsigned short* whi = (unsigned short*)d_ws;           // 512 KB
  unsigned short* wlo = whi + (size_t)HID_ * IN_;        // 512 KB

  snn_wsplit_kernel<<<dim3((HID_ * IN_ + 255) / 256), dim3(256), 0, stream>>>(
      W, whi, wlo, HID_ * IN_);

  const int smem_bytes = (2 * HT_ * WS_ + 2 * CH_ * XS_) * 2 + CH_ * ZS_ * 4;  // 208384
  (void)hipFuncSetAttribute((const void*)snn_fused_kernel,
                            hipFuncAttributeMaxDynamicSharedMemorySize, smem_bytes);
  snn_fused_kernel<<<dim3(HID_ / HT_, B_), dim3(256), smem_bytes, stream>>>(
      x, whi, wlo, bias, thr, out);
}
